// CausalSelfAttention_56624848830618
// MI455X (gfx1250) — matrix operand order repaired and checked by reading, not compiled, not run on hardware
//
#include <hip/hip_runtime.h>

#define Bsz 2
#define Tsz 2048
#define Csz 1024
#define Hn  16
#define Dh  64

typedef __attribute__((ext_vector_type(16))) _Float16 v16h;
typedef __attribute__((ext_vector_type(8)))  _Float16 v8h;
typedef __attribute__((ext_vector_type(8)))  float    v8f;
typedef __attribute__((ext_vector_type(4)))  float    v4f;

union AF { v16h v; v8h h[2]; };

// A fragment (16x32 f16) from row-major LDS tile (origin = base, leading dim ld halves).
// Lane l<16 holds row l, K={0..7,16..23}; lane l>=16 holds row l-16, K={8..15,24..31}.
__device__ __forceinline__ v16h load_frag_a(const _Float16* base, int lane, int ld) {
    int r = lane & 15, hi = lane >> 4;
    const _Float16* p = base + r * ld + hi * 8;
    AF a;
    a.h[0] = *(const v8h*)(p);
    a.h[1] = *(const v8h*)(p + 16);
    return a.v;
}

// B fragment (32x16 f16) from LDS tile stored TRANSPOSED, i.e. [N][K] row-major.
// Lane l holds column N=l&15, K = (l<16 ? {0..7,16..23} : {8..15,24..31}).
__device__ __forceinline__ v16h load_frag_b(const _Float16* base, int lane, int ld) {
    int c = lane & 15, hi = lane >> 4;
    const _Float16* p = base + c * ld + hi * 8;
    AF b;
    b.h[0] = *(const v8h*)(p);
    b.h[1] = *(const v8h*)(p + 16);
    return b.v;
}

// dst[N][K] (f16) = src[K][N] (f32)
__global__ void transpose_cvt_kernel(const float* __restrict__ src,
                                     _Float16* __restrict__ dst, int K, int N) {
    size_t i = (size_t)blockIdx.x * blockDim.x + threadIdx.x;
    size_t total = (size_t)K * N;
    if (i >= total) return;
    int n = (int)(i / K), k = (int)(i % K);
    dst[i] = (_Float16)src[(size_t)k * N + n];
}

// ---------------- GEMM 1: X[4096,1024](f32) @ Wqkv -> scatter Q,K (BHTD) and Vt (BHDT), f16
// 256 threads = 8 waves; block tile 128x128; wave tile 32x64; BK=32; double-buffered LDS.
__global__ __launch_bounds__(256) void qkv_gemm_kernel(
    const float* __restrict__ X, const _Float16* __restrict__ Wt,
    const float* __restrict__ bias, _Float16* __restrict__ Qb,
    _Float16* __restrict__ Kb, _Float16* __restrict__ Vtb) {
    const int LDT = 40;
    __shared__ _Float16 As[2][128 * LDT];
    __shared__ _Float16 Bs[2][128 * LDT];
    int tid = threadIdx.x, lane = tid & 31, wid = tid >> 5;
    int m0 = blockIdx.x * 128, n0 = blockIdx.y * 128;
    int wm = (wid >> 1) * 32, wn = (wid & 1) * 64;
    v8f acc[2][4] = {};
    v4f areg[4];
    v8h breg[2];

    auto gload = [&](int k0) {
#pragma unroll
        for (int it = 0; it < 4; ++it) {
            int idx = tid * 4 + it * 1024;
            int r = idx >> 5, c = idx & 31;
            areg[it] = *(const v4f*)(X + (size_t)(m0 + r) * Csz + k0 + c);
        }
#pragma unroll
        for (int it = 0; it < 2; ++it) {
            int idx = tid * 8 + it * 2048;
            int r = idx >> 5, c = idx & 31;
            breg[it] = *(const v8h*)(Wt + (size_t)(n0 + r) * Csz + k0 + c);
        }
    };
    auto sstore = [&](int buf) {
#pragma unroll
        for (int it = 0; it < 4; ++it) {
            int idx = tid * 4 + it * 1024;
            int r = idx >> 5, c = idx & 31;
            _Float16* d = As[buf] + r * LDT + c;
            d[0] = (_Float16)areg[it].x; d[1] = (_Float16)areg[it].y;
            d[2] = (_Float16)areg[it].z; d[3] = (_Float16)areg[it].w;
        }
#pragma unroll
        for (int it = 0; it < 2; ++it) {
            int idx = tid * 8 + it * 2048;
            int r = idx >> 5, c = idx & 31;
            *(v8h*)(Bs[buf] + r * LDT + c) = breg[it];
        }
    };

    gload(0);
    sstore(0);
    __syncthreads();
    const int KSTEPS = Csz / 32;
    for (int ks = 0; ks < KSTEPS; ++ks) {
        int cur = ks & 1, nxt = cur ^ 1;
        if (ks + 1 < KSTEPS) gload((ks + 1) * 32);  // overlap global loads with WMMAs
        v16h af[2], bf[4];
        af[0] = load_frag_a(As[cur] + wm * LDT, lane, LDT);
        af[1] = load_frag_a(As[cur] + (wm + 16) * LDT, lane, LDT);
#pragma unroll
        for (int j = 0; j < 4; ++j)
            bf[j] = load_frag_b(Bs[cur] + (wn + j * 16) * LDT, lane, LDT);
#pragma unroll
        for (int i = 0; i < 2; ++i)
#pragma unroll
            for (int j = 0; j < 4; ++j)
                acc[i][j] = __builtin_amdgcn_wmma_f32_16x16x32_f16(
                    false, af[i], false, bf[j], (short)0, acc[i][j], false, false);
        if (ks + 1 < KSTEPS) sstore(nxt);
        __syncthreads();
    }
    int hi = lane >> 4, cl = lane & 15;
#pragma unroll
    for (int mi = 0; mi < 2; ++mi)
#pragma unroll
        for (int ni = 0; ni < 4; ++ni)
            for (int g = 0; g < 8; ++g) {
                int row = m0 + wm + mi * 16 + hi * 8 + g;   // [0,4096)
                int col = n0 + wn + ni * 16 + cl;           // [0,3072)
                float val = acc[mi][ni][g] + bias[col];
                int which = col >> 10, cc = col & 1023;
                int hh = cc >> 6, dd = cc & 63;
                int bb = row >> 11, tt = row & (Tsz - 1);
                size_t bh = (size_t)bb * Hn + hh;
                if (which == 0)      Qb[(bh * Tsz + tt) * Dh + dd] = (_Float16)val;
                else if (which == 1) Kb[(bh * Tsz + tt) * Dh + dd] = (_Float16)val;
                else                 Vtb[(bh * Dh + dd) * Tsz + tt] = (_Float16)val;
            }
}

// ---------------- Flash attention: one block = (b,h, 64-query tile), 4 waves x 16 rows,
// double-buffered K / V^T staging.
__global__ __launch_bounds__(128) void attn_kernel(
    const _Float16* __restrict__ Qb, const _Float16* __restrict__ Kb,
    const _Float16* __restrict__ Vtb, _Float16* __restrict__ attb) {
    const int LD = 72;
    __shared__ _Float16 Qs[64 * LD];
    __shared__ _Float16 Ks[2][64 * LD];
    __shared__ _Float16 Vts[2][64 * LD];
    __shared__ _Float16 Ps[4 * 16 * LD];
    int tid = threadIdx.x, lane = tid & 31, wid = tid >> 5;
    int qblk = blockIdx.x & 31;   // T/64 = 32
    int bh   = blockIdx.x >> 5;   // b*H + h
    int qb = qblk * 64;
    const _Float16* Qg = Qb + (size_t)bh * Tsz * Dh;
    const _Float16* Kg = Kb + (size_t)bh * Tsz * Dh;
    const _Float16* Vg = Vtb + (size_t)bh * Dh * Tsz;

    v8h kreg[4], vreg[4];
    auto loadKV = [&](int kb) {
#pragma unroll
        for (int it = 0; it < 4; ++it) {
            int idx = tid * 8 + it * 1024;
            int r = idx >> 6, c = idx & 63;
            kreg[it] = *(const v8h*)(Kg + (size_t)(kb + r) * Dh + c);
            vreg[it] = *(const v8h*)(Vg + (size_t)r * Tsz + kb + c);
        }
    };
    auto storeKV = [&](int buf) {
#pragma unroll
        for (int it = 0; it < 4; ++it) {
            int idx = tid * 8 + it * 1024;
            int r = idx >> 6, c = idx & 63;
            *(v8h*)(Ks[buf] + r * LD + c)  = kreg[it];
            *(v8h*)(Vts[buf] + r * LD + c) = vreg[it];
        }
    };

    // stage 64x64 Q tile
#pragma unroll
    for (int it = 0; it < 4; ++it) {
        int idx = tid * 8 + it * 1024;
        int r = idx >> 6, c = idx & 63;
        *(v8h*)(Qs + r * LD + c) = *(const v8h*)(Qg + (size_t)(qb + r) * Dh + c);
    }
    loadKV(0);
    storeKV(0);
    __syncthreads();

    v16h aq[2];
    aq[0] = load_frag_a(Qs + wid * 16 * LD, lane, LD);       // d = 0..31
    aq[1] = load_frag_a(Qs + wid * 16 * LD + 32, lane, LD);  // d = 32..63
    float m_run[8], l_run[8];
    v8f o[4] = {};
    for (int g = 0; g < 8; ++g) { m_run[g] = -__builtin_inff(); l_run[g] = 0.f; }
    int hi = lane >> 4, cl = lane & 15;
    _Float16* Pw = Ps + wid * 16 * LD;

    int nblk = qb / 64 + 1;
    for (int ib = 0; ib < nblk; ++ib) {
        int kb = ib * 64;
        int cur = ib & 1, nxt = cur ^ 1;
        if (ib + 1 < nblk) loadKV(kb + 64);  // overlap with compute

        // S = (Q K^T) / 8 : 4 tiles of 16x16
        v8f s[4];
#pragma unroll
        for (int ni = 0; ni < 4; ++ni) {
            v8f z = {};
#pragma unroll
            for (int kk = 0; kk < 2; ++kk) {
                v16h bk = load_frag_b(Ks[cur] + (ni * 16) * LD + kk * 32, lane, LD);
                z = __builtin_amdgcn_wmma_f32_16x16x32_f16(false, aq[kk], false, bk,
                                                           (short)0, z, false, false);
            }
            for (int g = 0; g < 8; ++g) z[g] *= 0.125f;
            s[ni] = z;
        }
        if (kb == qb) {  // diagonal block: causal mask
#pragma unroll
            for (int ni = 0; ni < 4; ++ni)
                for (int g = 0; g < 8; ++g) {
                    int qrow = qb + wid * 16 + hi * 8 + g;
                    int kcol = kb + ni * 16 + cl;
                    if (kcol > qrow) s[ni][g] = -__builtin_inff();
                }
        }
        // online softmax (row = (g, lane-half); reduce across 16-lane halves)
        float rm[8], alpha[8], rs[8] = {};
        for (int g = 0; g < 8; ++g) {
            float v = fmaxf(fmaxf(s[0][g], s[1][g]), fmaxf(s[2][g], s[3][g]));
            for (int m = 1; m < 16; m <<= 1) v = fmaxf(v, __shfl_xor(v, m, 32));
            rm[g] = v;
        }
        for (int g = 0; g < 8; ++g) {
            float mn = fmaxf(m_run[g], rm[g]);
            alpha[g] = __expf(m_run[g] - mn);
            m_run[g] = mn;
        }
#pragma unroll
        for (int ni = 0; ni < 4; ++ni)
            for (int g = 0; g < 8; ++g) {
                float p = __expf(s[ni][g] - m_run[g]);
                rs[g] += p;
                Pw[(hi * 8 + g) * LD + ni * 16 + cl] = (_Float16)p;  // C-layout -> LDS
            }
        for (int g = 0; g < 8; ++g) {
            float v = rs[g];
            for (int m = 1; m < 16; m <<= 1) v += __shfl_xor(v, m, 32);
            l_run[g] = l_run[g] * alpha[g] + v;
        }
#pragma unroll
        for (int ni = 0; ni < 4; ++ni)
            for (int g = 0; g < 8; ++g) o[ni][g] *= alpha[g];
        // O += P @ V (same-wave LDS store->load kept in order by DScnt)
        v16h ap[2];
        ap[0] = load_frag_a(Pw, lane, LD);
        ap[1] = load_frag_a(Pw + 32, lane, LD);
#pragma unroll
        for (int ni = 0; ni < 4; ++ni)
#pragma unroll
            for (int kk = 0; kk < 2; ++kk) {
                v16h bv = load_frag_b(Vts[cur] + (ni * 16) * LD + kk * 32, lane, LD);
                o[ni] = __builtin_amdgcn_wmma_f32_16x16x32_f16(false, ap[kk], false, bv,
                                                               (short)0, o[ni], false, false);
            }
        if (ib + 1 < nblk) storeKV(nxt);
        __syncthreads();
    }
    int bidx = bh >> 4, hh = bh & 15;
#pragma unroll
    for (int ni = 0; ni < 4; ++ni)
        for (int g = 0; g < 8; ++g) {
            int qrow = qb + wid * 16 + hi * 8 + g;
            float val = o[ni][g] / l_run[g];
            attb[((size_t)bidx * Tsz + qrow) * Csz + hh * Dh + ni * 16 + cl] = (_Float16)val;
        }
}

// ---------------- GEMM 2: att[4096,1024] f16 @ Wproj + bias -> out f32
// Same 128x128 / 8-wave / double-buffered structure; A source already f16.
__global__ __launch_bounds__(256) void proj_gemm_kernel(
    const _Float16* __restrict__ Ain, const _Float16* __restrict__ Wt,
    const float* __restrict__ bias, float* __restrict__ out) {
    const int LDT = 40;
    __shared__ _Float16 As[2][128 * LDT];
    __shared__ _Float16 Bs[2][128 * LDT];
    int tid = threadIdx.x, lane = tid & 31, wid = tid >> 5;
    int m0 = blockIdx.x * 128, n0 = blockIdx.y * 128;
    int wm = (wid >> 1) * 32, wn = (wid & 1) * 64;
    v8f acc[2][4] = {};
    v8h areg[2], breg[2];

    auto gload = [&](int k0) {
#pragma unroll
        for (int it = 0; it < 2; ++it) {
            int idx = tid * 8 + it * 2048;
            int r = idx >> 5, c = idx & 31;
            areg[it] = *(const v8h*)(Ain + (size_t)(m0 + r) * Csz + k0 + c);
            breg[it] = *(const v8h*)(Wt + (size_t)(n0 + r) * Csz + k0 + c);
        }
    };
    auto sstore = [&](int buf) {
#pragma unroll
        for (int it = 0; it < 2; ++it) {
            int idx = tid * 8 + it * 2048;
            int r = idx >> 5, c = idx & 31;
            *(v8h*)(As[buf] + r * LDT + c) = areg[it];
            *(v8h*)(Bs[buf] + r * LDT + c) = breg[it];
        }
    };

    gload(0);
    sstore(0);
    __syncthreads();
    const int KSTEPS = Csz / 32;
    for (int ks = 0; ks < KSTEPS; ++ks) {
        int cur = ks & 1, nxt = cur ^ 1;
        if (ks + 1 < KSTEPS) gload((ks + 1) * 32);
        v16h af[2], bf[4];
        af[0] = load_frag_a(As[cur] + wm * LDT, lane, LDT);
        af[1] = load_frag_a(As[cur] + (wm + 16) * LDT, lane, LDT);
#pragma unroll
        for (int j = 0; j < 4; ++j)
            bf[j] = load_frag_b(Bs[cur] + (wn + j * 16) * LDT, lane, LDT);
#pragma unroll
        for (int i = 0; i < 2; ++i)
#pragma unroll
            for (int j = 0; j < 4; ++j)
                acc[i][j] = __builtin_amdgcn_wmma_f32_16x16x32_f16(
                    false, af[i], false, bf[j], (short)0, acc[i][j], false, false);
        if (ks + 1 < KSTEPS) sstore(nxt);
        __syncthreads();
    }
    int hi = lane >> 4, cl = lane & 15;
#pragma unroll
    for (int mi = 0; mi < 2; ++mi)
#pragma unroll
        for (int ni = 0; ni < 4; ++ni)
            for (int g = 0; g < 8; ++g) {
                int row = m0 + wm + mi * 16 + hi * 8 + g;
                int col = n0 + wn + ni * 16 + cl;
                out[(size_t)row * Csz + col] = acc[mi][ni][g] + bias[col];
            }
}

extern "C" void kernel_launch(void* const* d_in, const int* in_sizes, int n_in,
                              void* d_out, int out_size, void* d_ws, size_t ws_size,
                              hipStream_t stream) {
    const float* x      = (const float*)d_in[0];
    const float* w_qkv  = (const float*)d_in[1];
    const float* b_qkv  = (const float*)d_in[2];
    const float* w_proj = (const float*)d_in[3];
    const float* b_proj = (const float*)d_in[4];
    float* out = (float*)d_out;

    char* w = (char*)d_ws;
    _Float16* Wqkvt  = (_Float16*)w; w += (size_t)3 * Csz * Csz * 2;   // [3072][1024]
    _Float16* Wprojt = (_Float16*)w; w += (size_t)Csz * Csz * 2;       // [1024][1024]
    _Float16* Qb     = (_Float16*)w; w += (size_t)Bsz * Csz * Tsz * 2; // [B,H,T,D]
    _Float16* Kb     = (_Float16*)w; w += (size_t)Bsz * Csz * Tsz * 2; // [B,H,T,D]
    _Float16* Vtb    = (_Float16*)w; w += (size_t)Bsz * Csz * Tsz * 2; // [B,H,D,T]
    _Float16* attb   = (_Float16*)w; w += (size_t)Bsz * Tsz * Csz * 2; // [B*T, C]

    transpose_cvt_kernel<<<(3 * Csz * Csz) / 256, 256, 0, stream>>>(w_qkv, Wqkvt, Csz, 3 * Csz);
    transpose_cvt_kernel<<<(Csz * Csz) / 256, 256, 0, stream>>>(w_proj, Wprojt, Csz, Csz);
    qkv_gemm_kernel<<<dim3((Bsz * Tsz) / 128, (3 * Csz) / 128), 256, 0, stream>>>(
        x, Wqkvt, b_qkv, Qb, Kb, Vtb);
    attn_kernel<<<Bsz * Hn * (Tsz / 64), 128, 0, stream>>>(Qb, Kb, Vtb, attb);
    proj_gemm_kernel<<<dim3((Bsz * Tsz) / 128, Csz / 128), 256, 0, stream>>>(
        attb, Wprojt, b_proj, out);
}
